// myAdvTFRNNModel_51376398795364
// MI455X (gfx1250) — matrix operand order repaired and checked by reading, not compiled, not run on hardware
//
#include <hip/hip_runtime.h>
#include <hip/hip_bf16.h>

typedef __attribute__((ext_vector_type(16))) _Float16 v16h;
typedef __attribute__((ext_vector_type(8)))  float    v8f;

#define VOCAB 10
#define EMB   32
#define HID   64
#define NG    256   // 4*HID gate width
#define TLEN  2048
#define ROWS  16    // batch rows per workgroup

static __device__ __forceinline__ v8f wmma_f16(v16h a, v16h b, v8f c) {
  // D = A(16x32 f16) * B(32x16 f16) + C(16x16 f32)
  return __builtin_amdgcn_wmma_f32_16x16x32_f16(false, a, false, b, (short)0, c,
                                                false, false);
}

// ---- fast transcendentals on the serial critical path ----
// CDNA5 has a hardware v_tanh_f32 (TRANS32 pipe). Probe both builtin
// spellings; fall back to a branchless exp-based form if neither exists.
static __device__ __forceinline__ float fast_tanhf(float x) {
#if __has_builtin(__builtin_amdgcn_tanhf)
  return __builtin_amdgcn_tanhf(x);
#elif __has_builtin(__builtin_amdgcn_tanh_f32)
  return __builtin_amdgcn_tanh_f32(x);
#else
  float xc = fminf(fmaxf(x, -20.0f), 20.0f);   // keep exp in range, branchless
  float e  = __expf(-2.0f * xc);
  return (1.0f - e) / (1.0f + e);
#endif
}

static __device__ __forceinline__ float fast_sigmoidf(float x) {
  // sigmoid(x) = 0.5*tanh(x/2) + 0.5  -> one TRANS op + one FMA, no branches
  return fmaf(0.5f, fast_tanhf(0.5f * x), 0.5f);
}

// A fragment (16x32, f16) from LDS row-major [16][stride].
// Contiguous k order used here (B matches): lane -> m = lane%16; element i -> k = kofs + 16*(lane/16) + i
static __device__ __forceinline__ v16h load_afrag(const _Float16* base, int stride,
                                                  int lm, int lh, int kofs) {
  v16h a;
#pragma unroll
  for (int i = 0; i < 16; ++i) {
    int k = kofs + 16 * lh + i;
    a[i] = base[lm * stride + k];
  }
  return a;
}

// B fragment (32x16, f16) from global f32 row-major [K][ldn], fixed column.
// Contiguous k order used here (A matches): lane -> n = lane%16; element i -> k = kbase + 16*(lane/16) + i
static __device__ __forceinline__ v16h load_bfrag_f32(const float* __restrict__ W,
                                                      int ldn, int kbase, int col,
                                                      int lh) {
  v16h b;
#pragma unroll
  for (int i = 0; i < 16; ++i) {
    int k = kbase + 16 * lh + i;
    b[i] = (_Float16)W[k * ldn + col];
  }
  return b;
}

__global__ __launch_bounds__(256, 1)
void lstm_mlp_fused(const int* __restrict__ num1, const int* __restrict__ num2,
                    const float* __restrict__ E,  const float* __restrict__ Wx,
                    const float* __restrict__ Wh, const float* __restrict__ bg,
                    const float* __restrict__ W1, const float* __restrict__ b1,
                    const float* __restrict__ W2, const float* __restrict__ b2,
                    const float* __restrict__ W3, const float* __restrict__ b3,
                    float* __restrict__ out)
{
  __shared__ float     sXZ[VOCAB * NG];     // 10 KB: precomputed E@Wx + b LUT
  __shared__ _Float16  sH[2][ROWS][HID];    // running hidden state (f16)
  __shared__ _Float16  sA1[ROWS][64];       // MLP stage-1 activations
  __shared__ _Float16  sA2[ROWS][32];       // MLP stage-2 activations
  __shared__ int       sTok[2][ROWS];       // this step's tokens
  __shared__ float     sB1[64], sB2[32], sB3[16];

  const int tid  = threadIdx.x;
  const int wave = tid >> 5;
  const int lane = tid & 31;
  const int lm   = lane & 15;   // n (or m for A-frags)
  const int lh   = lane >> 4;   // half-wave group
  const int b0   = blockIdx.x * ROWS;

  // ---- one-time: fold embedding + input projection into a 10x256 LUT ----
  for (int idx = tid; idx < VOCAB * NG; idx += 256) {
    int v = idx / NG, g = idx % NG;
    float acc = bg[g];
#pragma unroll
    for (int e = 0; e < EMB; ++e) acc += E[v * EMB + e] * Wx[e * NG + g];
    sXZ[idx] = acc;
  }
  for (int i = tid; i < 2 * ROWS * HID; i += 256) ((_Float16*)sH)[i] = (_Float16)0.0f;
  if (tid < 64) sB1[tid] = b1[tid];
  if (tid < 32) sB2[tid] = b2[tid];
  if (tid < 16) sB3[tid] = (tid < VOCAB) ? b3[tid] : 0.0f;

  // ---- loop-invariant weight B-fragments, held in VGPRs ----
  const int s  = wave >> 2;   // LSTM stream (0: num1, 1: num2)
  const int hb = wave & 3;    // 16-wide hidden-unit block

  v16h whf[4][2];             // all waves: Wh tiles for gates i,f,g,o
#pragma unroll
  for (int q = 0; q < 4; ++q)
#pragma unroll
    for (int ks = 0; ks < 2; ++ks)
      whf[q][ks] = load_bfrag_f32(Wh, NG, 32 * ks, q * HID + 16 * hb + lm, lh);

  v16h w1f[4];                // waves 0-3: W1 [128,64], N tile = wave
  if (wave < 4)
#pragma unroll
    for (int ks = 0; ks < 4; ++ks)
      w1f[ks] = load_bfrag_f32(W1, 64, 32 * ks, 16 * wave + lm, lh);

  v16h w2f[2];                // waves 4-5: W2 [64,32]
  if (wave == 4 || wave == 5)
#pragma unroll
    for (int ks = 0; ks < 2; ++ks)
      w2f[ks] = load_bfrag_f32(W2, 32, 32 * ks, 16 * (wave - 4) + lm, lh);

  v16h w3f;                   // wave 6: W3 [32,10] padded to N=16
  if (wave == 6) {
#pragma unroll
    for (int i = 0; i < 16; ++i) {
      int k = 16 * lh + i;
      w3f[i] = (lm < VOCAB) ? (_Float16)W3[k * VOCAB + lm] : (_Float16)0.0f;
    }
  }

  v8f c_state = {};           // cell state: (m = r + 8*lh, hid = 16*hb + lm)

  __syncthreads();

  for (int t = 0; t < TLEN; ++t) {
    // stage this step's tokens; prefetch the token stream ahead
    if (tid < 32) {
      const int* np = (tid & 16) ? num2 : num1;
      int m = tid & 15;
      sTok[tid >> 4][m] = np[(b0 + m) * TLEN + t];
      if ((t & 15) == 0 && t + 64 < TLEN)
        __builtin_prefetch(&np[(b0 + m) * TLEN + t + 64], 0, 3);
    }
    // h_{t-1} A-fragments (read before barrier; written after next barrier)
    v16h ha0 = load_afrag(&sH[s][0][0], HID, lm, lh, 0);
    v16h ha1 = load_afrag(&sH[s][0][0], HID, lm, lh, 32);
    __syncthreads();

    // z = LUT[token] + h @ Wh  (4 gate tiles, K=64 -> 2 chained WMMAs each)
    v8f z[4];
#pragma unroll
    for (int q = 0; q < 4; ++q) {
      v8f acc = {};
      acc  = wmma_f16(ha0, whf[q][0], acc);
      z[q] = wmma_f16(ha1, whf[q][1], acc);
    }
    const int colb = 16 * hb + lm;  // hidden unit within each gate
#pragma unroll
    for (int r = 0; r < 8; ++r) {
      int m = r + 8 * lh;
      const float* xz = &sXZ[sTok[s][m] * NG];
      float iv = fast_sigmoidf(z[0][r] + xz[0 * HID + colb]);
      float fv = fast_sigmoidf(z[1][r] + xz[1 * HID + colb]);
      float gv = fast_tanhf   (z[2][r] + xz[2 * HID + colb]);
      float ov = fast_sigmoidf(z[3][r] + xz[3 * HID + colb]);
      float cn = fv * c_state[r] + iv * gv;
      c_state[r] = cn;
      sH[s][m][colb] = (_Float16)(ov * fast_tanhf(cn));
    }
    __syncthreads();

    // MLP stage 1: relu(concat(h1,h2)[16,128] @ W1 + b1)  (waves 0-3)
    if (wave < 4) {
      v8f acc = {};
#pragma unroll
      for (int ks = 0; ks < 4; ++ks) {
        v16h a = load_afrag(&sH[ks >> 1][0][0], HID, lm, lh, 32 * (ks & 1));
        acc = wmma_f16(a, w1f[ks], acc);
      }
      float bias = sB1[16 * wave + lm];
#pragma unroll
      for (int r = 0; r < 8; ++r)
        sA1[r + 8 * lh][16 * wave + lm] = (_Float16)fmaxf(acc[r] + bias, 0.0f);
    }
    __syncthreads();

    // MLP stage 2: relu([16,64] @ W2 + b2)  (waves 4-5)
    if (wave == 4 || wave == 5) {
      v8f acc = {};
#pragma unroll
      for (int ks = 0; ks < 2; ++ks) {
        v16h a = load_afrag(&sA1[0][0], 64, lm, lh, 32 * ks);
        acc = wmma_f16(a, w2f[ks], acc);
      }
      float bias = sB2[16 * (wave - 4) + lm];
#pragma unroll
      for (int r = 0; r < 8; ++r)
        sA2[r + 8 * lh][16 * (wave - 4) + lm] = (_Float16)fmaxf(acc[r] + bias, 0.0f);
    }
    __syncthreads();

    // MLP stage 3: [16,32] @ W3 + b3 -> logits  (wave 6)
    if (wave == 6) {
      v16h a = load_afrag(&sA2[0][0], 32, lm, lh, 0);
      v8f acc = {};
      acc = wmma_f16(a, w3f, acc);
      if (lm < VOCAB) {
        float bias = sB3[lm];
#pragma unroll
        for (int r = 0; r < 8; ++r) {
          size_t m = (size_t)(b0 + r + 8 * lh);
          out[(m * TLEN + t) * VOCAB + lm] = acc[r] + bias;
        }
      }
    }
    __syncthreads();
  }
}

extern "C" void kernel_launch(void* const* d_in, const int* in_sizes, int n_in,
                              void* d_out, int out_size, void* d_ws, size_t ws_size,
                              hipStream_t stream) {
  const int*   num1 = (const int*)d_in[0];
  const int*   num2 = (const int*)d_in[1];
  const float* E    = (const float*)d_in[2];
  const float* Wx   = (const float*)d_in[3];
  const float* Wh   = (const float*)d_in[4];
  const float* bg   = (const float*)d_in[5];
  const float* W1   = (const float*)d_in[6];
  const float* b1   = (const float*)d_in[7];
  const float* W2   = (const float*)d_in[8];
  const float* b2   = (const float*)d_in[9];
  const float* W3   = (const float*)d_in[10];
  const float* b3   = (const float*)d_in[11];
  float* out = (float*)d_out;

  // 256 batch rows / 16 rows per workgroup = 16 workgroups, 8 wave32 each.
  lstm_mlp_fused<<<16, 256, 0, stream>>>(num1, num2, E, Wx, Wh, bg,
                                         W1, b1, W2, b2, W3, b3, out);
}
